// VectorQuantizer_78451872629101
// MI455X (gfx1250) — matrix operand order repaired and checked by reading, not compiled, not run on hardware
//
#include <hip/hip_runtime.h>
#include <hip/hip_bf16.h>
#include <math.h>

// Problem constants (from the reference)
#define BQ    32
#define DQ    256
#define HQ    32
#define WQ    32
#define KQ    1024
#define HW    (HQ * WQ)          // 1024
#define NTOK  (BQ * HW)          // 32768 tokens
#define NELEM (NTOK * DQ)        // 8388608 elements of Zq
#define BETAQ 0.25f

typedef __bf16 v8bf  __attribute__((ext_vector_type(8)));
typedef __bf16 v16bf __attribute__((ext_vector_type(16)));
typedef float  v8f   __attribute__((ext_vector_type(8)));

union VB16 { v16bf v; v8bf h[2]; };

// ---------------------------------------------------------------------------
// Workspace layout (bytes):
//   [0,      4096)   code_norm : 1024 f32   (||E_k||^2)
//   [4096,   528384) Ebf       : 1024*256 bf16
//   [528384, 659456) idx       : 32768 i32
//   [659456, 663552) counts    : 1024 u32
//   [663552, 665600) partials  : 512 f32    (per-block sum of (Zq-Ze)^2)
// ---------------------------------------------------------------------------

// ---- Kernel 0: E -> bf16, per-code squared norms, zero histogram ----------
__global__ __launch_bounds__(256) void vq_prep(const float* __restrict__ E,
                                               float* __restrict__ code_norm,
                                               __bf16* __restrict__ Ebf,
                                               unsigned int* __restrict__ counts) {
  __shared__ float red[256];
  const int k = blockIdx.x;      // code row
  const int t = threadIdx.x;     // d
  float e = E[k * DQ + t];
  Ebf[k * DQ + t] = (__bf16)e;
  red[t] = e * e;
  __syncthreads();
  for (int s = 128; s > 0; s >>= 1) {
    if (t < s) red[t] += red[t + s];
    __syncthreads();
  }
  if (t == 0) {
    code_norm[k] = red[0];
    counts[k] = 0u;              // must be re-zeroed every launch
  }
}

// ---- Kernel 1: fused distance matmul + argmin (WMMA bf16 hi/lo split) -----
// Block = 128 threads (4 waves). Block owns 16 tokens vs all 1024 codes.
// Each wave: 16 code-tiles (interleaved) x 8 k-steps x 2 WMMAs = 256 WMMAs.
__global__ __launch_bounds__(128) void vq_argmin(const float* __restrict__ inp,
                                                 const __bf16* __restrict__ Ebf,
                                                 const float* __restrict__ code_norm,
                                                 int* __restrict__ idx_out) {
  // Padded rows (264 halfs = 528 B): 16B-aligned fragments, no bank conflicts.
  __shared__ __align__(16) __bf16 Ah[16 * 264];
  __shared__ __align__(16) __bf16 Al[16 * 264];
  __shared__ float rv[4][16];
  __shared__ int   ri[4][16];

  const int t   = threadIdx.x;
  const int n0  = blockIdx.x * 16;       // first token of tile (16 | 1024 => same b)
  const int b   = n0 >> 10;
  const int hw0 = n0 & (HW - 1);
  const float* src = inp + (size_t)b * DQ * HW + hw0;

  // Stage A tile: 16 tokens x 256 d, f32 -> (bf16 hi, bf16 lo).
  // Thread t loads token (t&15), d = (t>>4) + 8j  (coalesced groups of 16).
  {
    const int tok = t & 15;
    const int d0  = t >> 4;
#pragma unroll
    for (int j = 0; j < 32; ++j) {
      int d = d0 + j * 8;
      float x = src[(size_t)d * HW + tok];
      __bf16 h = (__bf16)x;
      __bf16 l = (__bf16)(x - (float)h);
      Ah[tok * 264 + d] = h;
      Al[tok * 264 + d] = l;
    }
  }
  __syncthreads();

  const int lane = t & 31;
  const int wv   = t >> 5;             // wave id 0..3
  const int row  = lane & 15;          // A row (token) / B column (code) in tile
  const int aoff = (lane >> 4) * 8;    // A: K {0-7,16-23} vs {8-15,24-31}
  const int boff = (lane >> 4) * 8;    // B: K {0-7,16-23} vs {8-15,24-31}

  float minv[8];
  int   mini[8];
#pragma unroll
  for (int r = 0; r < 8; ++r) { minv[r] = 3.4e38f; mini[r] = 0; }

  for (int tile = wv; tile < 64; tile += 4) {
    const int c0 = tile * 16;
    const __bf16* Brow = Ebf + (size_t)(c0 + row) * DQ;
    v8f acc = {};
#pragma unroll
    for (int ks = 0; ks < 8; ++ks) {
      const int k0 = ks * 32;
      VB16 a, al, bb;
      a.h[0]  = *(const v8bf*)(&Ah[row * 264 + k0 + aoff]);
      a.h[1]  = *(const v8bf*)(&Ah[row * 264 + k0 + 16 + aoff]);
      al.h[0] = *(const v8bf*)(&Al[row * 264 + k0 + aoff]);
      al.h[1] = *(const v8bf*)(&Al[row * 264 + k0 + 16 + aoff]);
      bb.h[0] = *(const v8bf*)(Brow + k0 + boff);
      bb.h[1] = *(const v8bf*)(Brow + k0 + 16 + boff);
      acc = __builtin_amdgcn_wmma_f32_16x16x32_bf16(false, a.v,  false, bb.v,
                                                    (short)0, acc, false, false);
      acc = __builtin_amdgcn_wmma_f32_16x16x32_bf16(false, al.v, false, bb.v,
                                                    (short)0, acc, false, false);
    }
    // distance discriminant: ||e_k||^2 - 2 z.e_k  (||z||^2 constant per row)
    const float cn  = code_norm[c0 + row];
    const int   col = c0 + row;
#pragma unroll
    for (int r = 0; r < 8; ++r) {
      // C layout: VGPR r -> row r (lanes 0-15) / row r+8 (lanes 16-31); lane = column.
      float v = cn - 2.0f * acc[r];
      if (v < minv[r]) { minv[r] = v; mini[r] = col; }
    }
  }

  // Cross-lane argmin within each 16-lane half (rows 0-7 live in lanes 0-15,
  // rows 8-15 in lanes 16-31). xor masks 1,2,4,8 stay within the half.
#pragma unroll
  for (int m = 8; m >= 1; m >>= 1) {
#pragma unroll
    for (int r = 0; r < 8; ++r) {
      float ov = __shfl_xor(minv[r], m, 32);
      int   oi = __shfl_xor(mini[r], m, 32);
      if (ov < minv[r] || (ov == minv[r] && oi < mini[r])) {
        minv[r] = ov; mini[r] = oi;
      }
    }
  }
  if (lane == 0) {
#pragma unroll
    for (int r = 0; r < 8; ++r) { rv[wv][r] = minv[r]; ri[wv][r] = mini[r]; }
  }
  if (lane == 16) {
#pragma unroll
    for (int r = 0; r < 8; ++r) { rv[wv][8 + r] = minv[r]; ri[wv][8 + r] = mini[r]; }
  }
  __syncthreads();
  if (t < 16) {   // combine the 4 waves (disjoint code ranges), lowest-index tiebreak
    float bv = rv[0][t]; int bi = ri[0][t];
#pragma unroll
    for (int w = 1; w < 4; ++w) {
      float v = rv[w][t]; int i = ri[w][t];
      if (v < bv || (v == bv && i < bi)) { bv = v; bi = i; }
    }
    idx_out[n0 + t] = bi;
  }
}

// ---- Kernel 2: gather Zq (straight-through), MSE partials, histogram ------
// Block = 256 threads handles 64 tokens (same b). thread: hw = t&63, d = t>>6 + 4j.
__global__ __launch_bounds__(256) void vq_gather(const float* __restrict__ inp,
                                                 const float* __restrict__ E,
                                                 const int* __restrict__ idx,
                                                 float* __restrict__ outZq,
                                                 unsigned int* __restrict__ counts,
                                                 float* __restrict__ partials) {
  __shared__ float red[256];
  const int t  = threadIdx.x;
  const int n0 = blockIdx.x * 64;
  const int b  = n0 >> 10;
  const int n  = n0 + (t & 63);
  const int k  = idx[n];
  if (t < 64) atomicAdd(&counts[k], 1u);   // integer atomics => deterministic

  const size_t base = (size_t)b * DQ * HW + (size_t)(n & (HW - 1));
  float acc = 0.0f;
  for (int d = (t >> 6); d < DQ; d += 4) {
    float q = E[(size_t)k * DQ + d];
    float z = inp[base + (size_t)d * HW];
    outZq[base + (size_t)d * HW] = z + (q - z);   // straight-through forward
    float diff = q - z;
    acc += diff * diff;
  }
  red[t] = acc;
  __syncthreads();
  for (int s = 128; s > 0; s >>= 1) {
    if (t < s) red[t] += red[t + s];
    __syncthreads();
  }
  if (t == 0) partials[blockIdx.x] = red[0];   // no float atomics
}

// ---- Kernel 3: entropy -> words_used, mse -> losses, write scalars --------
__global__ __launch_bounds__(1024) void vq_final(const unsigned int* __restrict__ counts,
                                                 const float* __restrict__ partials,
                                                 float* __restrict__ out) {
  __shared__ float red[1024];
  __shared__ float ent_s;
  const int t = threadIdx.x;

  float p = (float)counts[t] * (1.0f / (float)NTOK);
  red[t] = -p * log2f(p + 1e-10f);
  __syncthreads();
  for (int s = 512; s > 0; s >>= 1) {
    if (t < s) red[t] += red[t + s];
    __syncthreads();
  }
  if (t == 0) ent_s = red[0];
  __syncthreads();

  red[t] = (t < 512) ? partials[t] : 0.0f;
  __syncthreads();
  for (int s = 512; s > 0; s >>= 1) {
    if (t < s) red[t] += red[t + s];
    __syncthreads();
  }
  if (t == 0) {
    float mse = red[0] * (1.0f / (float)NELEM);   // e_latent == q_latent numerically
    out[0]          = (1.0f + BETAQ) * mse;       // loss
    out[1 + NELEM]  = mse;                        // e_latent_loss
    out[2 + NELEM]  = mse;                        // q_latent_loss
    out[3 + NELEM]  = exp2f(ent_s);               // words_used
  }
}

// ---------------------------------------------------------------------------
extern "C" void kernel_launch(void* const* d_in, const int* in_sizes, int n_in,
                              void* d_out, int out_size, void* d_ws, size_t ws_size,
                              hipStream_t stream) {
  const float* inp = (const float*)d_in[0];   // [32,256,32,32] f32
  const float* E   = (const float*)d_in[1];   // [1024,256]     f32
  float* out = (float*)d_out;                 // [1 + NELEM + 3] f32

  char* ws = (char*)d_ws;
  float*        code_norm = (float*)(ws);
  __bf16*       Ebf       = (__bf16*)(ws + 4096);
  int*          idx       = (int*)(ws + 4096 + 524288);
  unsigned int* counts    = (unsigned int*)(ws + 4096 + 524288 + 131072);
  float*        partials  = (float*)(ws + 4096 + 524288 + 131072 + 4096);

  vq_prep  <<<KQ,        256, 0, stream>>>(E, code_norm, Ebf, counts);
  vq_argmin<<<NTOK / 16, 128, 0, stream>>>(inp, Ebf, code_norm, idx);
  vq_gather<<<NTOK / 64, 256, 0, stream>>>(inp, E, idx, out + 1, counts, partials);
  vq_final <<<1,        1024, 0, stream>>>(counts, partials, out);
}
